// Attention_68685116997866
// MI455X (gfx1250) — matrix operand order repaired and checked by reading, not compiled, not run on hardware
//
#include <hip/hip_runtime.h>

// ---------------------------------------------------------------------------
// Shapes
// ---------------------------------------------------------------------------
constexpr int B_  = 8;
constexpr int C_  = 128;
constexpr int C2_ = 256;
constexpr int N_  = 32768;
constexpr float INV_SCALE = 1.0f / 16.0f;   // 1/sqrt(2C)

// ws layout (in floats)
constexpr size_t WS_G   = 0;                                   // B*C2*C2 (atomic accum)
constexpr size_t WS_S   = WS_G   + (size_t)B_ * C2_ * C2_;     // B*C2    (atomic accum)
constexpr size_t WS_H   = WS_S   + (size_t)B_ * C2_;           // B*C2
constexpr size_t WS_P   = WS_H   + (size_t)B_ * C2_;           // B*C2*C2
constexpr size_t WS_E   = WS_P   + (size_t)B_ * C2_ * C2_;     // B*C2*C2
constexpr size_t WS_E1  = WS_E   + (size_t)B_ * C2_ * C2_;     // B*C2*C2
constexpr size_t WS_E2  = WS_E1  + (size_t)B_ * C2_ * C2_;     // B*C2*C
constexpr size_t WS_ATT = WS_E2  + (size_t)B_ * C2_ * C_;      // B*C2*C
constexpr size_t WS_M1  = WS_ATT + (size_t)B_ * C2_ * C_;      // B*C*C
constexpr size_t WS_M2  = WS_M1  + (size_t)B_ * C_ * C_;       // B*C*C
constexpr size_t WS_R   = WS_M2  + (size_t)B_ * C_ * C_;       // B*C

// ---------------------------------------------------------------------------
// WMMA types / helpers
// ---------------------------------------------------------------------------
typedef __attribute__((ext_vector_type(16))) __bf16 v16bf;
typedef __attribute__((ext_vector_type(8)))  __bf16 v8bf;
typedef __attribute__((ext_vector_type(8)))  float  v8f;

__device__ __forceinline__ __bf16 tobf(float f) {
  return static_cast<__bf16>(f);   // RNE fptrunc -> native v_cvt_pk_bf16_f32
}

__device__ __forceinline__ v16bf catv(v8bf lo, v8bf hi) {
  v16bf r;
#pragma unroll
  for (int i = 0; i < 8; ++i) { r[i] = lo[i]; r[i + 8] = hi[i]; }
  return r;
}

// A-frag (16x32 bf16): source stored k-minor, element (m,k) at base[m*S + k].
// ISA layout: lanes 0-15 -> K 0..7 then 16..23; lanes 16-31 -> +8.
__device__ __forceinline__ v16bf load_afrag(const __bf16* base, int S, int lane) {
  int half = lane >> 4, m = lane & 15;
  const __bf16* p = base + m * S;
  v8bf lo = *reinterpret_cast<const v8bf*>(p + 8 * half);
  v8bf hi = *reinterpret_cast<const v8bf*>(p + 16 + 8 * half);
  return catv(lo, hi);
}

// B-frag (32x16 bf16): source stored k-minor per column, element (k,n) at base[n*S + k].
// ISA layout: lanes 0-15 -> K 0..7 then 16..23; lanes 16-31 -> +8, N = lane&15.
__device__ __forceinline__ v16bf load_bfrag(const __bf16* base, int S, int lane) {
  int half = lane >> 4, n = lane & 15;
  const __bf16* p = base + n * S + 8 * half;
  v8bf lo = *reinterpret_cast<const v8bf*>(p);
  v8bf hi = *reinterpret_cast<const v8bf*>(p + 16);
  return catv(lo, hi);
}

// pack 8 floats -> v8bf
__device__ __forceinline__ v8bf pack8(float4 f0, float4 f1) {
  v8bf v;
  v[0] = tobf(f0.x); v[1] = tobf(f0.y); v[2] = tobf(f0.z); v[3] = tobf(f0.w);
  v[4] = tobf(f1.x); v[5] = tobf(f1.y); v[6] = tobf(f1.z); v[7] = tobf(f1.w);
  return v;
}

// ---------------------------------------------------------------------------
// Kernel 0: zero the atomic accumulators (G, S)
// ---------------------------------------------------------------------------
__global__ void zero_kernel(float* p, size_t n) {
  size_t i = (size_t)blockIdx.x * blockDim.x + threadIdx.x;
  if (i < n) p[i] = 0.f;
}

// ---------------------------------------------------------------------------
// Kernel 1: G[b] = Z Zᵀ  (Z = [y;x], [256, N]) + row sums s[b] = Z·1.
// Symmetric Gram: one staged 256-row panel serves both A and B sides, so x,y
// are read exactly once (256 MB total, roofline-optimal).
// Double-buffered LDS, compute-then-stage order (one barrier per stage).
// sched_barrier fences every 4 tiles so the scheduler cannot preload all 16
// B-fragments (128 VGPRs) on top of the 128 accumulator VGPRs -> no spills.
// ---------------------------------------------------------------------------
constexpr int NSPLIT = 32;
constexpr int CHUNK  = N_ / NSPLIT;   // 1024
constexpr int KC     = 64;            // staged k per iteration
constexpr int NSTG   = CHUNK / KC;    // 16
constexpr int LDA    = 88;            // padded LDS stride (16B aligned, bank-spread)

__global__ __launch_bounds__(512) void gram_kernel(
    const float* __restrict__ x, const float* __restrict__ y,
    float* __restrict__ G, float* __restrict__ S) {
  __shared__ __bf16 smZ[2 * 256 * LDA];   // 90 KB, double-buffered

  int wg    = blockIdx.x;
  int chunk = wg % NSPLIT;
  int b     = wg / NSPLIT;

  int t = threadIdx.x;
  int w = t >> 5, lane = t & 31;
  int row = t >> 1, seg = (t & 1) * 32;   // 2 threads per Z row, 32 floats each

  const float* zr = (row < C_) ? (y + ((size_t)(b * C_ + row)) * N_)
                               : (x + ((size_t)(b * C_ + (row - C_))) * N_);
  int nbase = chunk * CHUNK;

  v8f acc[16];
#pragma unroll
  for (int i = 0; i < 16; ++i) acc[i] = (v8f)(0.0f);

  float rs = 0.f;

  auto stage = [&](int stg, int bufi) {
    const float4* p = reinterpret_cast<const float4*>(zr + nbase + stg * KC + seg);
    __builtin_prefetch(zr + nbase + (stg + 2) * KC + seg, 0, 3);  // 2 ahead, WGP scope
    __bf16* dst = &smZ[bufi * (256 * LDA) + row * LDA + seg];
#pragma unroll
    for (int q = 0; q < 4; ++q) {
      float4 f0 = p[2 * q], f1 = p[2 * q + 1];
      rs += f0.x + f0.y + f0.z + f0.w + f1.x + f1.y + f1.z + f1.w;
      *reinterpret_cast<v8bf*>(dst + q * 8) = pack8(f0, f1);
    }
  };

  stage(0, 0);
  for (int stg = 0; stg < NSTG; ++stg) {
    int cur = stg & 1;
    __syncthreads();                        // buf[cur] visible; prev compute done
    const __bf16* Zp = &smZ[cur * (256 * LDA)];
#pragma unroll
    for (int ks = 0; ks < 2; ++ks) {
      v16bf a = load_afrag(Zp + (w * 16) * LDA + ks * 32, LDA, lane);
#pragma unroll
      for (int g = 0; g < 4; ++g) {
        v16bf bm[4];
#pragma unroll
        for (int i = 0; i < 4; ++i)
          bm[i] = load_bfrag(Zp + ((g * 4 + i) * 16) * LDA + ks * 32, LDA, lane);
#pragma unroll
        for (int i = 0; i < 4; ++i)
          acc[g * 4 + i] = __builtin_amdgcn_wmma_f32_16x16x32_bf16(
              false, a, false, bm[i], (short)0, acc[g * 4 + i], false, false);
        __builtin_amdgcn_sched_barrier(0);   // cap in-flight B-frags -> no spills
      }
    }
    if (stg + 1 < NSTG) stage(stg + 1, cur ^ 1);   // behind other waves' compute
  }

  int half = lane >> 4, ln = lane & 15;
#pragma unroll
  for (int ct = 0; ct < 16; ++ct) {
    int gj = ct * 16 + ln;
#pragma unroll
    for (int v = 0; v < 8; ++v) {
      int gi = w * 16 + half * 8 + v;
      atomicAdd(&G[((size_t)(b * C2_) + gi) * C2_ + gj], acc[ct][v]);
    }
  }
  atomicAdd(&S[b * C2_ + row], rs);
}

// ---------------------------------------------------------------------------
// Tiny per-batch chain (0.7 GFLOP total, spread over B*256 blocks: VALU f32)
// ---------------------------------------------------------------------------
__global__ __launch_bounds__(256) void k2_h(const float* __restrict__ S,
                                            const float* __restrict__ q_w,
                                            const float* __restrict__ k_w,
                                            float* __restrict__ H) {
  int b = blockIdx.x, i = threadIdx.x;
  const float* wrow = (i < C_) ? (q_w + (size_t)i * C_) : (k_w + (size_t)(i - C_) * C_);
  const float* srow = S + b * C2_ + ((i < C_) ? 0 : C_);
  float acc = 0.f;
  for (int k = 0; k < C_; ++k) acc += wrow[k] * srow[k];
  H[b * C2_ + i] = acc;
}

__global__ __launch_bounds__(256) void k2_P(const float* __restrict__ G,
                                            const float* __restrict__ q_w,
                                            const float* __restrict__ k_w,
                                            float* __restrict__ P) {
  int blk = blockIdx.x, b = blk >> 8, krow = blk & 255, j = threadIdx.x;
  const float* grow = G + ((size_t)(b * C2_) + krow) * C2_ + ((j < C_) ? 0 : C_);
  const float* wrow = (j < C_) ? (q_w + (size_t)j * C_) : (k_w + (size_t)(j - C_) * C_);
  float acc = 0.f;
  for (int l = 0; l < C_; ++l) acc += grow[l] * wrow[l];
  P[((size_t)(b * C2_) + krow) * C2_ + j] = acc;
}

__global__ __launch_bounds__(256) void k2_E(const float* __restrict__ P,
                                            const float* __restrict__ q_w,
                                            const float* __restrict__ k_w,
                                            const float* __restrict__ q_b,
                                            const float* __restrict__ k_b,
                                            const float* __restrict__ H,
                                            float* __restrict__ E) {
  int blk = blockIdx.x, b = blk >> 8, i = blk & 255, j = threadIdx.x;
  const float* wrow = (i < C_) ? (q_w + (size_t)i * C_) : (k_w + (size_t)(i - C_) * C_);
  int koff = (i < C_) ? 0 : C_;
  const float* pb = P + (size_t)(b * C2_) * C2_;
  float acc = 0.f;
  for (int k = 0; k < C_; ++k) acc += wrow[k] * pb[(size_t)(koff + k) * C2_ + j];
  float bci = (i < C_) ? q_b[i] : k_b[i - C_];
  float bcj = (j < C_) ? q_b[j] : k_b[j - C_];
  float hi = H[b * C2_ + i], hj = H[b * C2_ + j];
  acc += hi * bcj + bci * hj + (float)N_ * bci * bcj;
  E[((size_t)(b * C2_) + i) * C2_ + j] = acc * INV_SCALE;
}

__global__ __launch_bounds__(256) void k2_mlp1(const float* __restrict__ E,
                                               const float* __restrict__ t1_w,
                                               const float* __restrict__ t1_b,
                                               float* __restrict__ e1) {
  int blk = blockIdx.x, b = blk >> 8, i = blk & 255, j = threadIdx.x;
  const float* erow = E + ((size_t)(b * C2_) + i) * C2_;
  const float* wrow = t1_w + (size_t)j * C2_;
  float acc = t1_b[j];
  for (int d = 0; d < C2_; ++d) acc += erow[d] * wrow[d];
  e1[((size_t)(b * C2_) + i) * C2_ + j] = fmaxf(acc, 0.f);
}

__global__ __launch_bounds__(128) void k2_mlp2(const float* __restrict__ e1,
                                               const float* __restrict__ t2_w,
                                               const float* __restrict__ t2_b,
                                               float* __restrict__ e2) {
  int blk = blockIdx.x, b = blk >> 8, i = blk & 255, j = threadIdx.x;
  const float* erow = e1 + ((size_t)(b * C2_) + i) * C2_;
  const float* wrow = t2_w + (size_t)j * C2_;
  float acc = t2_b[j];
  for (int d = 0; d < C2_; ++d) acc += erow[d] * wrow[d];
  e2[((size_t)(b * C2_) + i) * C_ + j] = fmaxf(acc, 0.f);
}

__global__ __launch_bounds__(128) void k2_softmax(const float* __restrict__ e2,
                                                  float* __restrict__ att) {
  __shared__ float red[128];
  int blk = blockIdx.x, j = threadIdx.x;
  float v = e2[(size_t)blk * C_ + j];
  red[j] = v; __syncthreads();
  for (int s = 64; s > 0; s >>= 1) { if (j < s) red[j] = fmaxf(red[j], red[j + s]); __syncthreads(); }
  float m = red[0]; __syncthreads();
  float ex = __expf(v - m);
  red[j] = ex; __syncthreads();
  for (int s = 64; s > 0; s >>= 1) { if (j < s) red[j] += red[j + s]; __syncthreads(); }
  att[(size_t)blk * C_ + j] = ex / red[0];
}

__global__ __launch_bounds__(128) void k2_fold(const float* __restrict__ att,
                                               const float* __restrict__ v1_w,
                                               const float* __restrict__ v1_b,
                                               const float* __restrict__ v2_w,
                                               const float* __restrict__ v2_b,
                                               float* __restrict__ M1,
                                               float* __restrict__ M2,
                                               float* __restrict__ R) {
  int blk = blockIdx.x, b = blk >> 7, k = blk & 127, c = threadIdx.x;
  const float* atop = att + (size_t)(b * C2_) * C_;
  const float* abot = atop + (size_t)C_ * C_;
  float a1 = 0.f, a2 = 0.f;
  for (int d = 0; d < C_; ++d) {
    a1 += v2_w[(size_t)d * C_ + k] * atop[(size_t)d * C_ + c];
    a2 += v1_w[(size_t)d * C_ + k] * abot[(size_t)d * C_ + c];
  }
  M1[((size_t)(b * C_) + k) * C_ + c] = a1;
  M2[((size_t)(b * C_) + k) * C_ + c] = a2;
  if (k == 0) {
    float r = 0.f;
    for (int d = 0; d < C_; ++d)
      r += v2_b[d] * atop[(size_t)d * C_ + c] + v1_b[d] * abot[(size_t)d * C_ + c];
    R[b * C_ + c] = r;
  }
}

// ---------------------------------------------------------------------------
// Kernel 3: out[b] = M1ᵀ·y[b] + M2ᵀ·x[b] + r·1ᵀ   (bf16 WMMA, LDS-staged)
// Both Mᵀ panels staged up front; streamed x/y tiles double-buffered with
// compute-then-stage order. Grid: B * (N/128), 8 waves/WG.
// ---------------------------------------------------------------------------
constexpr int LDM = 152;  // Mᵀ LDS stride (bf16 elems)
constexpr int LDB = 40;   // streamed-tile LDS stride (bf16 elems)

__global__ __launch_bounds__(256) void out_kernel(
    const float* __restrict__ x, const float* __restrict__ y,
    const float* __restrict__ M1, const float* __restrict__ M2,
    const float* __restrict__ R, float* __restrict__ out) {
  __shared__ __bf16 smM[2][128 * LDM];  // M1ᵀ, M2ᵀ: element (c,k) at smM[ph][c*LDM + k]
  __shared__ __bf16 smB[2][128 * LDB];  // streamed tile: element (k,n) at smB[buf][n*LDB + k]

  int nb = blockIdx.x & 255;
  int b  = blockIdx.x >> 8;
  int n0 = nb * 128;
  int t = threadIdx.x, w = t >> 5, lane = t & 31;

  v8f acc[8];
#pragma unroll
  for (int i = 0; i < 8; ++i) acc[i] = (v8f)(0.0f);

  // stage M1ᵀ and M2ᵀ as bf16 (one-time)
  {
    int kM = t >> 1, csM = (t & 1) * 64;
#pragma unroll
    for (int ph = 0; ph < 2; ++ph) {
      const float* Msrc = (ph == 0) ? M1 : M2;
      const float4* p4 = reinterpret_cast<const float4*>(
          Msrc + ((size_t)(b * C_ + kM)) * C_ + csM);
#pragma unroll
      for (int q = 0; q < 16; ++q) {
        float4 f = p4[q];
        smM[ph][(csM + 4 * q + 0) * LDM + kM] = tobf(f.x);
        smM[ph][(csM + 4 * q + 1) * LDM + kM] = tobf(f.y);
        smM[ph][(csM + 4 * q + 2) * LDM + kM] = tobf(f.z);
        smM[ph][(csM + 4 * q + 3) * LDM + kM] = tobf(f.w);
      }
    }
  }

  int kr = t >> 3, cs8 = (t & 7) * 16;   // streamed-tile staging assignment

  auto stageB = [&](int u, int bufi) {   // u = ph*4 + ks
    const float* src = (u < 4) ? y : x;
    int k0 = (u & 3) * 32;
    const float4* p4 = reinterpret_cast<const float4*>(
        src + ((size_t)(b * C_ + k0 + kr)) * N_ + n0 + cs8);
    int up = (u + 2 < 8) ? u + 2 : 7;    // clamped 2-ahead prefetch
    const float* psrc = (up < 4) ? y : x;
    __builtin_prefetch(psrc + ((size_t)(b * C_ + (up & 3) * 32 + kr)) * N_ + n0 + cs8, 0, 3);
    __bf16* dst = smB[bufi];
#pragma unroll
    for (int q = 0; q < 4; ++q) {
      float4 f = p4[q];
      dst[(cs8 + 4 * q + 0) * LDB + kr] = tobf(f.x);
      dst[(cs8 + 4 * q + 1) * LDB + kr] = tobf(f.y);
      dst[(cs8 + 4 * q + 2) * LDB + kr] = tobf(f.z);
      dst[(cs8 + 4 * q + 3) * LDB + kr] = tobf(f.w);
    }
  };

  stageB(0, 0);
  for (int u = 0; u < 8; ++u) {
    int cur = u & 1;
    __syncthreads();                      // smM (u==0) and smB[cur] visible
    int ph = u >> 2, k0 = (u & 3) * 32;
    v16bf a = load_afrag(&smM[ph][(w * 16) * LDM + k0], LDM, lane);
    v16bf bm[8];
#pragma unroll
    for (int nt = 0; nt < 8; ++nt)
      bm[nt] = load_bfrag(&smB[cur][(nt * 16) * LDB], LDB, lane);
#pragma unroll
    for (int nt = 0; nt < 8; ++nt)
      acc[nt] = __builtin_amdgcn_wmma_f32_16x16x32_bf16(
          false, a, false, bm[nt], (short)0, acc[nt], false, false);
    if (u + 1 < 8) stageB(u + 1, cur ^ 1);   // behind other waves' compute
  }

  int half = lane >> 4, ln = lane & 15;
#pragma unroll
  for (int nt = 0; nt < 8; ++nt) {
#pragma unroll
    for (int v = 0; v < 8; ++v) {
      int c = w * 16 + half * 8 + v;
      out[((size_t)(b * C_ + c)) * N_ + (n0 + nt * 16 + ln)] =
          acc[nt][v] + R[b * C_ + c];
    }
  }
}

// ---------------------------------------------------------------------------
// Launch
// ---------------------------------------------------------------------------
extern "C" void kernel_launch(void* const* d_in, const int* in_sizes, int n_in,
                              void* d_out, int out_size, void* d_ws, size_t ws_size,
                              hipStream_t stream) {
  (void)in_sizes; (void)n_in; (void)out_size; (void)ws_size;
  const float* x    = (const float*)d_in[0];
  const float* y    = (const float*)d_in[1];
  const float* q_w  = (const float*)d_in[2];
  const float* q_b  = (const float*)d_in[3];
  const float* k_w  = (const float*)d_in[4];
  const float* k_b  = (const float*)d_in[5];
  const float* v1_w = (const float*)d_in[6];
  const float* v1_b = (const float*)d_in[7];
  const float* v2_w = (const float*)d_in[8];
  const float* v2_b = (const float*)d_in[9];
  const float* t1_w = (const float*)d_in[10];
  const float* t1_b = (const float*)d_in[11];
  const float* t2_w = (const float*)d_in[12];
  const float* t2_b = (const float*)d_in[13];
  float* out = (float*)d_out;
  float* ws  = (float*)d_ws;

  float* G  = ws + WS_G;  float* S  = ws + WS_S;  float* H   = ws + WS_H;
  float* P  = ws + WS_P;  float* E  = ws + WS_E;  float* e1  = ws + WS_E1;
  float* e2 = ws + WS_E2; float* att = ws + WS_ATT;
  float* M1 = ws + WS_M1; float* M2 = ws + WS_M2; float* R   = ws + WS_R;

  size_t nz = WS_H;  // zero the atomic accumulators (G and S are contiguous)
  zero_kernel<<<dim3((unsigned)((nz + 255) / 256)), 256, 0, stream>>>(ws, nz);
  gram_kernel<<<dim3(B_ * NSPLIT), 512, 0, stream>>>(x, y, G, S);
  k2_h      <<<dim3(B_),        256, 0, stream>>>(S, q_w, k_w, H);
  k2_P      <<<dim3(B_ * C2_),  256, 0, stream>>>(G, q_w, k_w, P);
  k2_E      <<<dim3(B_ * C2_),  256, 0, stream>>>(P, q_w, k_w, q_b, k_b, H, E);
  k2_mlp1   <<<dim3(B_ * C2_),  256, 0, stream>>>(E, t1_w, t1_b, e1);
  k2_mlp2   <<<dim3(B_ * C2_),  128, 0, stream>>>(e1, t2_w, t2_b, e2);
  k2_softmax<<<dim3(B_ * C2_),  128, 0, stream>>>(e2, att);
  k2_fold   <<<dim3(B_ * C_),   128, 0, stream>>>(att, v1_w, v1_b, v2_w, v2_b, M1, M2, R);
  out_kernel<<<dim3(B_ * (N_ / 128)), 256, 0, stream>>>(x, y, M1, M2, R, out);
}
